// TransformerBlock_87548613361686
// MI455X (gfx1250) — matrix operand order repaired and checked by reading, not compiled, not run on hardware
//
#include <hip/hip_runtime.h>
#include <hip/hip_bf16.h>

typedef __attribute__((ext_vector_type(16))) _Float16 v16h;
typedef __attribute__((ext_vector_type(8)))  _Float16 v8h;
typedef __attribute__((ext_vector_type(8)))  float    v8f;

// ---------------------------------------------------------------------------
// WMMA helpers (CDNA5 16x16x32 f16 -> f32)
// ---------------------------------------------------------------------------
__device__ __forceinline__ v8f wmma_f16(v16h a, v16h b, v8f c) {
  return __builtin_amdgcn_wmma_f32_16x16x32_f16(
      /*neg_a=*/false, a, /*neg_b=*/false, b,
      /*c_mod=*/(short)0, c, /*reuse_a=*/false, /*reuse_b=*/false);
}

// A-matrix 16x32 f16 fragment from row-major [M][ld] memory.
// ISA layout: lanes 0-15 (m=lane): elems 0..7 = K0..7, 8..15 = K16..23
//             lanes 16-31 (m=lane-16): elems 0..7 = K8..15, 8..15 = K24..31
__device__ __forceinline__ v16h load_fragA(const _Float16* base, int ld, int lane) {
  const int m  = lane & 15;
  const int k0 = (lane & 16) ? 8 : 0;
  const _Float16* p = base + (size_t)m * ld + k0;
  v8h lo = *(const v8h*)(p);
  v8h hi = *(const v8h*)(p + 16);
  v16h f;
#pragma unroll
  for (int i = 0; i < 8; ++i) { f[i] = lo[i]; f[i + 8] = hi[i]; }
  return f;
}

// B-matrix 32x16 f16 fragment from row-major Bt[N][ld] (i.e. B transposed).
// ISA layout: lanes 0-15 (n=lane): elems 0..7 = K0..7, 8..15 = K16..23; lanes 16-31: K8..15, K24..31.
__device__ __forceinline__ v16h load_fragB(const _Float16* baseT, int ld, int lane) {
  const int n  = lane & 15;
  const int kb = (lane & 16) ? 8 : 0;
  const _Float16* p = baseT + (size_t)n * ld + kb;
  v8h lo = *(const v8h*)(p);
  v8h hi = *(const v8h*)(p + 16);
  v16h f;
#pragma unroll
  for (int i = 0; i < 8; ++i) { f[i] = lo[i]; f[i + 8] = hi[i]; }
  return f;
}

// GELU (tanh form), rewritten as v * sigmoid(2u): one v_exp_f32 instead of tanhf
__device__ __forceinline__ float gelu_f(float v) {
  const float u = 0.7978845608028654f * (v + 0.044715f * v * v * v);
  return v / (1.0f + __expf(-2.0f * u));
}

// ---------------------------------------------------------------------------
// Weight transpose + f32->f16 convert: wt[out][in] = (f16) w[in][out]
// ---------------------------------------------------------------------------
__global__ __launch_bounds__(256) void k_transpose_cvt(
    const float* __restrict__ w, _Float16* __restrict__ wt, int IN, int OUT) {
  size_t idx = (size_t)blockIdx.x * 256 + threadIdx.x;
  size_t total = (size_t)IN * OUT;
  if (idx >= total) return;
  size_t o = idx / (size_t)IN;
  size_t i = idx % (size_t)IN;
  wt[idx] = (_Float16)w[i * (size_t)OUT + o];
}

// ---------------------------------------------------------------------------
// LayerNorm (faithful: unbiased std, eps added to std), f32 in -> f16 out
// ---------------------------------------------------------------------------
__global__ __launch_bounds__(256) void k_layernorm(
    const float* __restrict__ x, const float* __restrict__ scale,
    const float* __restrict__ shift, _Float16* __restrict__ out) {
  __shared__ float red[256];
  const int row = blockIdx.x;
  const int t = threadIdx.x;
  const float* xr = x + (size_t)row * 1024;
  float v[4];
  float s = 0.f;
#pragma unroll
  for (int i = 0; i < 4; ++i) { v[i] = xr[t + 256 * i]; s += v[i]; }
  red[t] = s;
  __syncthreads();
  for (int off = 128; off > 0; off >>= 1) {
    if (t < off) red[t] += red[t + off];
    __syncthreads();
  }
  const float mean = red[0] * (1.0f / 1024.0f);
  __syncthreads();
  float sq = 0.f;
#pragma unroll
  for (int i = 0; i < 4; ++i) { float d = v[i] - mean; sq += d * d; }
  red[t] = sq;
  __syncthreads();
  for (int off = 128; off > 0; off >>= 1) {
    if (t < off) red[t] += red[t + off];
    __syncthreads();
  }
  const float var = red[0] * (1.0f / 1023.0f);  // ddof=1
  const float inv = 1.0f / (sqrtf(var) + 1e-5f);
  _Float16* orow = out + (size_t)row * 1024;
#pragma unroll
  for (int i = 0; i < 4; ++i) {
    int c = t + 256 * i;
    orow[c] = (_Float16)(scale[c] * ((v[i] - mean) * inv) + shift[c]);
  }
}

// ---------------------------------------------------------------------------
// GEMM: C[M,N] = A[M,K] @ Bt[N,K]^T + bias, all-f16 inputs, f32 accum.
// MODE 0: f16 out         MODE 1: f16 out with GELU       MODE 2: f32 out + residual
// Block = 256 threads = 8 waves, wave tile 32(M)x64(N), block tile 128x128.
// Per K-step: 2 A-frags + 4 B-frags feed 8 v_wmma (0.75 fragment loads / WMMA);
// K-loop unrolled x2 so two load groups are in flight against the WMMA chain.
// ---------------------------------------------------------------------------
template <int MODE>
__global__ __launch_bounds__(256) void k_gemm(
    const _Float16* __restrict__ A, const _Float16* __restrict__ Bt,
    const float* __restrict__ bias, void* __restrict__ Cout,
    const float* __restrict__ residual, int M, int N, int K) {
  const int lane = threadIdx.x & 31;
  const int w    = threadIdx.x >> 5;
  const int wm   = w >> 1, wn = w & 1;
  const int m0 = blockIdx.y * 128 + wm * 32;
  const int n0 = blockIdx.x * 128 + wn * 64;

  v8f c[2][4];
#pragma unroll
  for (int i = 0; i < 2; ++i)
#pragma unroll
    for (int j = 0; j < 4; ++j) c[i][j] = (v8f){};

  const _Float16* ap = A  + (size_t)m0 * K;
  const _Float16* bp = Bt + (size_t)n0 * K;

#pragma unroll 2
  for (int k = 0; k < K; k += 32) {
    // prefetch next iteration's cache lines (speculative; OOB harmlessly dropped)
    __builtin_prefetch(ap + k + 64, 0, 0);
    __builtin_prefetch(bp + k + 64, 0, 0);

    v16h af[2], bf[4];
#pragma unroll
    for (int i = 0; i < 2; ++i)
      af[i] = load_fragA(ap + (size_t)(i * 16) * K + k, K, lane);
#pragma unroll
    for (int j = 0; j < 4; ++j)
      bf[j] = load_fragB(bp + (size_t)(j * 16) * K + k, K, lane);
#pragma unroll
    for (int i = 0; i < 2; ++i)
#pragma unroll
      for (int j = 0; j < 4; ++j) c[i][j] = wmma_f16(af[i], bf[j], c[i][j]);
  }

  const int hh = lane >> 4;   // accum half: rows r (0) / r+8 (1)
  const int nl = lane & 15;
#pragma unroll
  for (int i = 0; i < 2; ++i) {
#pragma unroll
    for (int j = 0; j < 4; ++j) {
      const int mt = m0 + i * 16;
      const int col = n0 + j * 16 + nl;
      const float bv = bias[col];
#pragma unroll
      for (int r = 0; r < 8; ++r) {
        const int row = mt + r + (hh << 3);
        const size_t idx = (size_t)row * N + col;
        float v = c[i][j][r] + bv;
        if (MODE == 0) {
          ((_Float16*)Cout)[idx] = (_Float16)v;
        } else if (MODE == 1) {
          ((_Float16*)Cout)[idx] = (_Float16)gelu_f(v);
        } else {
          ((float*)Cout)[idx] = v + residual[idx];
        }
      }
    }
  }
}

// ---------------------------------------------------------------------------
// Flash attention (causal). Grid: (S/32, B*H). Block = 64 threads = 2 waves,
// each wave owns one 16-row query tile. Layouts: Q,K,V,ctx are f16 [B,S,EMB],
// head slice at column h*64. Softmax streaming with per-row running max/sum.
// ---------------------------------------------------------------------------
#define SEQLEN 2048
#define NHEADS 16
#define HDIM   64

__global__ __launch_bounds__(64) void k_attn(
    const _Float16* __restrict__ Q, const _Float16* __restrict__ Kf,
    const _Float16* __restrict__ V, _Float16* __restrict__ ctx) {
  __shared__ __align__(16) _Float16 vt_s[2][64 * 40];  // V^T per-wave: [d][kpos], ld=40
  __shared__ __align__(16) _Float16 pb_s[2][16 * 32];  // P per-wave: [q][kpos], ld=32

  const int lane = threadIdx.x & 31;
  const int wid  = threadIdx.x >> 5;
  _Float16* vt = vt_s[wid];
  _Float16* pb = pb_s[wid];

  const int q0 = blockIdx.x * 32 + wid * 16;
  const int bh = blockIdx.y;
  const int b  = bh >> 4;
  const int h  = bh & 15;
  const size_t rowbase = (size_t)b * SEQLEN;

  const _Float16* Qb = Q + (rowbase + q0) * 1024 + h * HDIM;
  const v16h aq0 = load_fragA(Qb, 1024, lane);        // d = 0..31
  const v16h aq1 = load_fragA(Qb + 32, 1024, lane);   // d = 32..63

  const int hh = lane >> 4;
  const int nl = lane & 15;
  float Mrun[8], Lrun[8];
#pragma unroll
  for (int r = 0; r < 8; ++r) { Mrun[r] = -1e30f; Lrun[r] = 0.0f; }
  v8f o0 = {}, o1 = {}, o2 = {}, o3 = {};

  const int nchunk = blockIdx.x + 1;  // causal: keys up to q0+15
  for (int kc = 0; kc < nchunk; ++kc) {
    const int kpos0 = kc * 32;

    // ---- scores: Q (16x64) @ K^T (64x32) -> two 16x16 tiles ----
    const _Float16* Kb = Kf + (rowbase + kpos0) * 1024 + h * HDIM;
    v16h b00 = load_fragB(Kb, 1024, lane);                 // keys 0..15, d 0..31
    v16h b01 = load_fragB(Kb + 32, 1024, lane);            // keys 0..15, d 32..63
    v16h b10 = load_fragB(Kb + (size_t)16 * 1024, 1024, lane);
    v16h b11 = load_fragB(Kb + (size_t)16 * 1024 + 32, 1024, lane);
    v8f s0 = {}, s1 = {};
    s0 = wmma_f16(aq0, b00, s0);
    s0 = wmma_f16(aq1, b01, s0);
    s1 = wmma_f16(aq0, b10, s1);
    s1 = wmma_f16(aq1, b11, s1);

    // ---- stage V chunk (32 x 64) transposed into LDS: vt[d][kpos] ----
    const _Float16* vrow = V + (rowbase + kpos0 + lane) * 1024 + h * HDIM;
#pragma unroll
    for (int j = 0; j < 8; ++j) {
      v8h ch = *(const v8h*)(vrow + j * 8);
#pragma unroll
      for (int e = 0; e < 8; ++e) vt[(j * 8 + e) * 40 + lane] = ch[e];
    }

    // ---- streaming causal softmax (scale 1/sqrt(64) = 0.125) ----
#pragma unroll
    for (int r = 0; r < 8; ++r) {
      const int qrow = q0 + r + (hh << 3);
      const int c0 = kpos0 + nl;
      const int c1 = kpos0 + 16 + nl;
      float x0 = (c0 <= qrow) ? s0[r] * 0.125f : -1e30f;
      float x1 = (c1 <= qrow) ? s1[r] * 0.125f : -1e30f;
      float rm = fmaxf(x0, x1);
      for (int off = 1; off < 16; off <<= 1) rm = fmaxf(rm, __shfl_xor(rm, off));
      const float mnew = fmaxf(Mrun[r], rm);
      const float p0 = __expf(x0 - mnew);
      const float p1 = __expf(x1 - mnew);
      float rs = p0 + p1;
      for (int off = 1; off < 16; off <<= 1) rs += __shfl_xor(rs, off);
      const float corr = __expf(Mrun[r] - mnew);
      Lrun[r] = Lrun[r] * corr + rs;
      Mrun[r] = mnew;
      o0[r] *= corr; o1[r] *= corr; o2[r] *= corr; o3[r] *= corr;
      const int prow = r + (hh << 3);
      pb[prow * 32 + nl]      = (_Float16)p0;
      pb[prow * 32 + 16 + nl] = (_Float16)p1;
    }

    // ---- O += P (16x32) @ V (32x64) ---- (same-wave LDS is in-order)
    v16h pa  = load_fragA(pb, 32, lane);
    v16h vb0 = load_fragB(vt + 0 * 16 * 40, 40, lane);
    v16h vb1 = load_fragB(vt + 1 * 16 * 40, 40, lane);
    v16h vb2 = load_fragB(vt + 2 * 16 * 40, 40, lane);
    v16h vb3 = load_fragB(vt + 3 * 16 * 40, 40, lane);
    o0 = wmma_f16(pa, vb0, o0);
    o1 = wmma_f16(pa, vb1, o1);
    o2 = wmma_f16(pa, vb2, o2);
    o3 = wmma_f16(pa, vb3, o3);
  }

  // ---- finalize: divide by softmax denominator and store f16 ctx ----
  _Float16* cb = ctx + (rowbase + q0) * 1024 + h * HDIM;
#pragma unroll
  for (int r = 0; r < 8; ++r) {
    const float inv = 1.0f / Lrun[r];
    const size_t row = (size_t)(r + (hh << 3)) * 1024;
    cb[row + nl]      = (_Float16)(o0[r] * inv);
    cb[row + 16 + nl] = (_Float16)(o1[r] * inv);
    cb[row + 32 + nl] = (_Float16)(o2[r] * inv);
    cb[row + 48 + nl] = (_Float16)(o3[r] * inv);
  }
}

// ---------------------------------------------------------------------------
// Host-side orchestration
// ---------------------------------------------------------------------------
extern "C" void kernel_launch(void* const* d_in, const int* in_sizes, int n_in,
                              void* d_out, int out_size, void* d_ws, size_t ws_size,
                              hipStream_t stream) {
  (void)in_sizes; (void)n_in; (void)out_size; (void)ws_size;

  constexpr int EMB = 1024, FF = 4096, BATCH = 2, SEQ = 2048;
  constexpr int MTOT = BATCH * SEQ;  // 4096

  const float* x         = (const float*)d_in[0];
  const float* ln1_scale = (const float*)d_in[1];
  const float* ln1_shift = (const float*)d_in[2];
  const float* ln2_scale = (const float*)d_in[3];
  const float* ln2_shift = (const float*)d_in[4];
  const float* wq = (const float*)d_in[5];  const float* bq = (const float*)d_in[6];
  const float* wk = (const float*)d_in[7];  const float* bk = (const float*)d_in[8];
  const float* wv = (const float*)d_in[9];  const float* bv = (const float*)d_in[10];
  const float* wo = (const float*)d_in[11]; const float* bo = (const float*)d_in[12];
  const float* w_fc   = (const float*)d_in[13]; const float* b_fc   = (const float*)d_in[14];
  const float* w_proj = (const float*)d_in[15]; const float* b_proj = (const float*)d_in[16];
  float* out = (float*)d_out;

  char* ws = (char*)d_ws;
  size_t off = 0;
  auto alloc = [&](size_t bytes) -> void* {
    void* p = ws + off;
    off += (bytes + 255) & ~(size_t)255;
    return p;
  };

  _Float16* wqT    = (_Float16*)alloc((size_t)EMB * EMB * 2);
  _Float16* wkT    = (_Float16*)alloc((size_t)EMB * EMB * 2);
  _Float16* wvT    = (_Float16*)alloc((size_t)EMB * EMB * 2);
  _Float16* woT    = (_Float16*)alloc((size_t)EMB * EMB * 2);
  _Float16* wfcT   = (_Float16*)alloc((size_t)EMB * FF * 2);   // [FF][EMB]
  _Float16* wprojT = (_Float16*)alloc((size_t)FF * EMB * 2);   // [EMB][FF]
  _Float16* xn     = (_Float16*)alloc((size_t)MTOT * EMB * 2); // LN out (reused for LN2)
  _Float16* Qf     = (_Float16*)alloc((size_t)MTOT * EMB * 2);
  _Float16* Kf     = (_Float16*)alloc((size_t)MTOT * EMB * 2);
  _Float16* Vf     = (_Float16*)alloc((size_t)MTOT * EMB * 2);
  _Float16* ctx    = (_Float16*)alloc((size_t)MTOT * EMB * 2);
  float*    hbuf   = (float*)   alloc((size_t)MTOT * EMB * 4);
  _Float16* aBuf   = Qf;  // FF activations (32 MB) overlay dead Q/K/V/ctx (4 x 8 MB)

  // 1) weight transpose+convert to f16 [N][K]
  k_transpose_cvt<<<(EMB * EMB + 255) / 256, 256, 0, stream>>>(wq, wqT, EMB, EMB);
  k_transpose_cvt<<<(EMB * EMB + 255) / 256, 256, 0, stream>>>(wk, wkT, EMB, EMB);
  k_transpose_cvt<<<(EMB * EMB + 255) / 256, 256, 0, stream>>>(wv, wvT, EMB, EMB);
  k_transpose_cvt<<<(EMB * EMB + 255) / 256, 256, 0, stream>>>(wo, woT, EMB, EMB);
  k_transpose_cvt<<<(EMB * FF + 255) / 256, 256, 0, stream>>>(w_fc, wfcT, EMB, FF);
  k_transpose_cvt<<<(FF * EMB + 255) / 256, 256, 0, stream>>>(w_proj, wprojT, FF, EMB);

  // 2) LN1
  k_layernorm<<<MTOT, 256, 0, stream>>>(x, ln1_scale, ln1_shift, xn);

  // 3) QKV projections
  {
    dim3 grid(EMB / 128, MTOT / 128);
    k_gemm<0><<<grid, 256, 0, stream>>>(xn, wqT, bq, (void*)Qf, nullptr, MTOT, EMB, EMB);
    k_gemm<0><<<grid, 256, 0, stream>>>(xn, wkT, bk, (void*)Kf, nullptr, MTOT, EMB, EMB);
    k_gemm<0><<<grid, 256, 0, stream>>>(xn, wvT, bv, (void*)Vf, nullptr, MTOT, EMB, EMB);
  }

  // 4) causal flash attention
  {
    dim3 grid(SEQ / 32, BATCH * 16);
    k_attn<<<grid, 64, 0, stream>>>(Qf, Kf, Vf, ctx);
  }

  // 5) output projection + residual -> h (f32)
  {
    dim3 grid(EMB / 128, MTOT / 128);
    k_gemm<2><<<grid, 256, 0, stream>>>(ctx, woT, bo, (void*)hbuf, x, MTOT, EMB, EMB);
  }

  // 6) LN2
  k_layernorm<<<MTOT, 256, 0, stream>>>(hbuf, ln2_scale, ln2_shift, xn);

  // 7) FC + GELU (f16)
  {
    dim3 grid(FF / 128, MTOT / 128);
    k_gemm<1><<<grid, 256, 0, stream>>>(xn, wfcT, b_fc, (void*)aBuf, nullptr, MTOT, FF, EMB);
  }

  // 8) down-projection + residual h -> out (f32)
  {
    dim3 grid(EMB / 128, MTOT / 128);
    k_gemm<2><<<grid, 256, 0, stream>>>(aBuf, wprojT, b_proj, (void*)out, hbuf, MTOT, EMB, FF);
  }
}
